// TopKSparseAttention_55783035240898
// MI455X (gfx1250) — matrix operand order repaired and checked by reading, compiled, not run on hardware
//
#include <hip/hip_runtime.h>
#include <hip/hip_bf16.h>

// ---------------------------------------------------------------------------
// Top-k sparse attention, B=2, C=256, H=W=64 (N=4096), k_keep=409.
//   wcvt_kernel : W fp32 -> bf16 (once, outside the GEMM hot loops)
//   proj_kernel : Q/K/V = x @ W^T + b (WMMA bf16, f32 accum); V stored
//                 transposed [b][c][m] so the attn kernel's B-frags are
//                 contiguous b128 loads.
//   attn_kernel : per 16-query tile: bf16 score strip in LDS (128KB ->
//                 2 blocks/WGP), TDM-staged Q tile, per-row top-k threshold
//                 by register-cached bisection + softmax in place, then
//                 dense P @ V with WMMA bf16.
//
// Fragment layouts per ISA 7.12.2 (16-bit A 16x32): lane's 16 elements are
// two contiguous 8-element K runs: K in [8h,8h+8) u [16+8h,16+8h+8), h=lane>>4.
// => row-major K-contiguous tiles give A-frag = 2x ds_load_b128 / b128 load.
// B 32x16: lane n = lane&15, element e -> same K as A (two 8-element runs).
// C/D f32 16x16: N = lane&15, M = r + 8h.
// ---------------------------------------------------------------------------

#define C_DIM 256
#define N_TOK 4096
#define KKEEP 409

typedef __attribute__((ext_vector_type(16))) __bf16           v16bf;
typedef __attribute__((ext_vector_type(8)))  float            v8f;
typedef __attribute__((ext_vector_type(16))) unsigned short   v16u;
typedef __attribute__((ext_vector_type(4)))  unsigned int     u32x4;
typedef __attribute__((ext_vector_type(4)))  int              i32x4;
typedef __attribute__((ext_vector_type(8)))  int              i32x8;

#if defined(__has_builtin)
#  if __has_builtin(__builtin_amdgcn_tensor_load_to_lds) && \
      __has_builtin(__builtin_amdgcn_s_wait_tensorcnt)
#    define USE_TDM 1
#  endif
#endif
#ifndef USE_TDM
#  define USE_TDM 0
#endif

union Frag {
    v16u           u;
    v16bf          b;
    uint4          q[2];
    unsigned short s[16];
};

// fp32 -> bf16 round-to-nearest-even
static __device__ __forceinline__ unsigned short f2bf(float f) {
    union { float f; unsigned u; } x; x.f = f;
    unsigned r = x.u + 0x7FFFu + ((x.u >> 16) & 1u);
    return (unsigned short)(r >> 16);
}
// u32 bit pattern -> float
static __device__ __forceinline__ float asf(unsigned u) {
    union { unsigned u; float f; } x; x.u = u;
    return x.f;
}

// ---------------------------------------------------------------------------
// Weight fp32 -> bf16, 256x256 row-major. grid*block covers 65536/4.
// ---------------------------------------------------------------------------
__global__ __launch_bounds__(256)
void wcvt_kernel(const float* __restrict__ W, unsigned short* __restrict__ out) {
    const int i = blockIdx.x * 256 + threadIdx.x;
    const float4 w = ((const float4*)W)[i];
    uint2 pk;
    pk.x = (unsigned)f2bf(w.x) | ((unsigned)f2bf(w.y) << 16);
    pk.y = (unsigned)f2bf(w.z) | ((unsigned)f2bf(w.w) << 16);
    ((uint2*)out)[i] = pk;
}

// ---------------------------------------------------------------------------
// Projection: out[token, o] = bf16( sum_c x[b,c,token] * W[o,c] + bias[o] )
// stored at out[b*N*C + token*ts + o*cs]  (Q/K: ts=C,cs=1; V: ts=1,cs=N).
// grid = B*256 tiles of 16 tokens, block = 128 (4 waves, 64 outs each).
// ---------------------------------------------------------------------------
__global__ __launch_bounds__(128)
void proj_kernel(const float* __restrict__ x, const unsigned short* __restrict__ Wbf,
                 const float* __restrict__ bias, unsigned short* __restrict__ out,
                 int ts, int cs) {
    __shared__ unsigned short xs[16 * C_DIM];   // row-major [t][c] bf16, 8KB

    const int blk  = blockIdx.x;
    const int b    = blk >> 8;
    const int t0   = (blk & 255) << 4;
    const int tid  = threadIdx.x;
    const int lane = tid & 31, wave = tid >> 5;
    const int half = lane >> 4, l16 = lane & 15;

    // stage x tile: read in [c][t] order (coalesced), scatter to [t][c] bf16
    const float* xb = x + (size_t)b * C_DIM * N_TOK + t0;
    for (int i = tid; i < 16 * C_DIM; i += 128) {
        const int c = i >> 4, t = i & 15;
        xs[t * C_DIM + c] = f2bf(xb[(size_t)c * N_TOK + t]);
    }
    __syncthreads();

    v8f acc[4] = {};
    const unsigned short* xrow = xs + l16 * C_DIM;

#pragma unroll
    for (int chunk = 0; chunk < 8; ++chunk) {
        Frag a;
        a.q[0] = *(const uint4*)(xrow + chunk * 32 + half * 8);
        a.q[1] = *(const uint4*)(xrow + chunk * 32 + 16 + half * 8);
#pragma unroll
        for (int sub = 0; sub < 4; ++sub) {
            const int o = wave * 64 + sub * 16 + l16;
            const uint4* wp =
                (const uint4*)(Wbf + (size_t)o * C_DIM + chunk * 32 + half * 8);
            Frag bm;
            bm.q[0] = wp[0];
            bm.q[1] = wp[2];
            acc[sub] = __builtin_amdgcn_wmma_f32_16x16x32_bf16(
                false, a.b, false, bm.b, (short)0, acc[sub], false, false);
        }
    }

#pragma unroll
    for (int sub = 0; sub < 4; ++sub) {
        const int o  = wave * 64 + sub * 16 + l16;
        const float bv = bias[o];
        unsigned short* ob = out + (size_t)b * N_TOK * C_DIM + (size_t)o * cs;
#pragma unroll
        for (int r = 0; r < 8; ++r) {
            const int token = t0 + r + half * 8;
            ob[(size_t)token * ts] = f2bf(acc[sub][r] + bv);
        }
    }
}

// ---------------------------------------------------------------------------
// Fused scores + top-k softmax + P@V.
// grid = B*256 (16-query tiles), block = 256 (8 waves).
// Dynamic LDS: score strip 16x4096 bf16 (128KB) + Q tile 16x256 bf16 (8KB).
// ---------------------------------------------------------------------------
__global__ __launch_bounds__(256)
void attn_kernel(const unsigned short* __restrict__ Qg,
                 const unsigned short* __restrict__ Kg,
                 const unsigned short* __restrict__ VT,
                 float* __restrict__ out) {
    extern __shared__ char smem[];
    unsigned short* sc = (unsigned short*)smem;                 // 16 x 4096 bf16
    unsigned short* qs = sc + 16 * N_TOK;                       // 16 x 256  bf16

    const int blk  = blockIdx.x;
    const int b    = blk >> 8;
    const int t0   = (blk & 255) << 4;
    const int tid  = threadIdx.x;
    const int lane = tid & 31, wave = tid >> 5;
    const int half = lane >> 4, l16 = lane & 15;

    // --- stage Q tile (16 x 256 bf16 = 8KB) into LDS ---
#if USE_TDM
    if (tid < 32) {
        // Tensor DMA descriptor: 2D tile 256(x) x 16(y), 2-byte elements,
        // global row stride 256, dest = qs (LDS byte offset 16*N_TOK*2).
        const unsigned long long ga =
            (unsigned long long)(Qg + ((size_t)b * N_TOK + t0) * C_DIM);
        u32x4 g0;
        g0[0] = 1u;                                      // count=1, user mode
        g0[1] = (unsigned)(16 * N_TOK * 2);              // lds_addr (bytes)
        g0[2] = (unsigned)(ga & 0xFFFFFFFFu);            // global_addr[31:0]
        g0[3] = (unsigned)((ga >> 32) & 0x01FFFFFFu)     // global_addr[56:32]
              | 0x80000000u;                             // type=2 ("image")
        i32x8 g1;
        g1[0] = 0x00010000;        // workgroup_mask=0, data_size=1 (2 bytes)
        g1[1] = 0x01000000;        // tensor_dim0 = 256  (bits 79:48)
        g1[2] = 0x00100000;        // tensor_dim1 = 16   (bits 111:80)
        g1[3] = 0x01000000;        // tile_dim0   = 256  (bits 127:112)
        g1[4] = 0x00000010;        // tile_dim1   = 16
        g1[5] = 256;               // tensor_dim0_stride = 256
        g1[6] = 0;
        g1[7] = 0;
        i32x4 g2 = {0, 0, 0, 0};
        i32x4 g3 = {0, 0, 0, 0};
#if defined(__clang_major__) && __clang_major__ >= 23
        i32x8 g4 = {0, 0, 0, 0, 0, 0, 0, 0};
        __builtin_amdgcn_tensor_load_to_lds(g0, g1, g2, g3, g4, 0);
#else
        __builtin_amdgcn_tensor_load_to_lds(g0, g1, g2, g3, 0);
#endif
        __builtin_amdgcn_s_wait_tensorcnt(0);
    }
    __syncthreads();
#else
    {
        const uint4* qgv = (const uint4*)(Qg + ((size_t)b * N_TOK + t0) * C_DIM);
        uint4*       qdv = (uint4*)qs;
        for (int i = tid; i < (16 * C_DIM) / 8; i += 256) qdv[i] = qgv[i];
    }
    __syncthreads();
#endif

    // --- hoist Q A-fragments: 2x ds_load_b128 per chunk ---
    Frag afr[8];
    {
        const unsigned short* qrow = qs + l16 * C_DIM;
#pragma unroll
        for (int chunk = 0; chunk < 8; ++chunk) {
            afr[chunk].q[0] = *(const uint4*)(qrow + chunk * 32 + half * 8);
            afr[chunk].q[1] = *(const uint4*)(qrow + chunk * 32 + 16 + half * 8);
        }
    }

    // --- phase B: S[16][4096] = bf16((Q K^T)/16), 16 keys per wave-iter ---
    for (int kt = wave; kt < 256; kt += 8) {
        const int key0 = kt << 4;
        const unsigned short* kb = Kg + ((size_t)b * N_TOK + key0 + l16) * C_DIM;
        if (kt + 8 < 256)
            __builtin_prefetch(Kg + ((size_t)b * N_TOK + ((kt + 8) << 4) + l16) * C_DIM, 0, 1);
        v8f acc = {};
#pragma unroll
        for (int chunk = 0; chunk < 8; ++chunk) {
            Frag bm;
            const uint4* kp = (const uint4*)(kb + chunk * 32 + half * 8);
            bm.q[0] = kp[0];
            bm.q[1] = kp[2];
            acc = __builtin_amdgcn_wmma_f32_16x16x32_bf16(
                false, afr[chunk].b, false, bm.b, (short)0, acc, false, false);
        }
#pragma unroll
        for (int r = 0; r < 8; ++r)
            sc[(r + half * 8) * N_TOK + key0 + l16] = f2bf(acc[r] * 0.0625f);
    }
    __syncthreads();

    // --- phase C: per-row top-409 threshold + softmax, register-cached ---
    for (int rr = 0; rr < 2; ++rr) {
        const int row = wave + rr * 8;
        unsigned short* srow = sc + (size_t)row * N_TOK;

        // each lane caches its 128 elements: 16x ds_load_b128, then all
        // bisection passes run on registers.
        uint4 cache[16];
#pragma unroll
        for (int g = 0; g < 16; ++g)
            cache[g] = *(const uint4*)(srow + (lane + g * 32) * 8);

        float mx = -3.4e38f, mn = 3.4e38f;
#pragma unroll
        for (int g = 0; g < 16; ++g) {
            const unsigned* cw = (const unsigned*)&cache[g];
#pragma unroll
            for (int w = 0; w < 4; ++w) {
                float v0 = asf(cw[w] << 16);
                float v1 = asf(cw[w] & 0xFFFF0000u);
                mx = fmaxf(mx, fmaxf(v0, v1));
                mn = fminf(mn, fminf(v0, v1));
            }
        }
#pragma unroll
        for (int off = 16; off >= 1; off >>= 1) {
            mx = fmaxf(mx, __shfl_xor(mx, off, 32));
            mn = fminf(mn, __shfl_xor(mn, off, 32));
        }

        float lo = mn, hi = mx;
        for (int it = 0; it < 16; ++it) {      // bf16 resolution: 16 steps
            const float mid = 0.5f * (lo + hi);
            int cnt = 0;
#pragma unroll
            for (int g = 0; g < 16; ++g) {
                const unsigned* cw = (const unsigned*)&cache[g];
#pragma unroll
                for (int w = 0; w < 4; ++w) {
                    cnt += (asf(cw[w] << 16) > mid) ? 1 : 0;
                    cnt += (asf(cw[w] & 0xFFFF0000u) > mid) ? 1 : 0;
                }
            }
#pragma unroll
            for (int off = 16; off >= 1; off >>= 1) cnt += __shfl_xor(cnt, off, 32);
            if (cnt > KKEEP) lo = mid; else hi = mid;
        }
        const float thresh = hi;

        float sum = 0.f;
#pragma unroll
        for (int g = 0; g < 16; ++g) {
            const unsigned* cw = (const unsigned*)&cache[g];
#pragma unroll
            for (int w = 0; w < 4; ++w) {
                float v0 = asf(cw[w] << 16);
                float v1 = asf(cw[w] & 0xFFFF0000u);
                if (v0 >= thresh) sum += __expf(v0 - mx);
                if (v1 >= thresh) sum += __expf(v1 - mx);
            }
        }
#pragma unroll
        for (int off = 16; off >= 1; off >>= 1) sum += __shfl_xor(sum, off, 32);
        const float inv = 1.0f / sum;

#pragma unroll
        for (int g = 0; g < 16; ++g) {
            const unsigned* cw = (const unsigned*)&cache[g];
            uint4 pk;
            unsigned* pw = (unsigned*)&pk;
#pragma unroll
            for (int w = 0; w < 4; ++w) {
                float v0 = asf(cw[w] << 16);
                float v1 = asf(cw[w] & 0xFFFF0000u);
                unsigned p0 = (v0 >= thresh) ? (unsigned)f2bf(__expf(v0 - mx) * inv) : 0u;
                unsigned p1 = (v1 >= thresh) ? (unsigned)f2bf(__expf(v1 - mx) * inv) : 0u;
                pw[w] = p0 | (p1 << 16);
            }
            *(uint4*)(srow + (lane + g * 32) * 8) = pk;
        }
    }
    __syncthreads();

    // --- phase D: out[16 x 256] = P[16 x 4096] @ V[4096 x 256] ---
    v8f acc2[2] = {};
    const int ct0 = wave * 2;
    const unsigned short* prow = sc + l16 * N_TOK;
    for (int mc = 0; mc < 128; ++mc) {
        const int m0 = mc << 5;
        Frag a;
        a.q[0] = *(const uint4*)(prow + m0 + half * 8);
        a.q[1] = *(const uint4*)(prow + m0 + 16 + half * 8);
#pragma unroll
        for (int t = 0; t < 2; ++t) {
            const int ch = ((ct0 + t) << 4) + l16;
            const uint4* vp =
                (const uint4*)(VT + ((size_t)b * C_DIM + ch) * N_TOK + m0 + half * 8);
            Frag bm;
            bm.q[0] = vp[0];
            bm.q[1] = vp[2];
            acc2[t] = __builtin_amdgcn_wmma_f32_16x16x32_bf16(
                false, a.b, false, bm.b, (short)0, acc2[t], false, false);
        }
    }
#pragma unroll
    for (int t = 0; t < 2; ++t) {
        const int ch = ((ct0 + t) << 4) + l16;
#pragma unroll
        for (int r = 0; r < 8; ++r) {
            const int token = t0 + r + half * 8;
            out[((size_t)b * C_DIM + ch) * N_TOK + token] = acc2[t][r];
        }
    }
}

// ---------------------------------------------------------------------------
extern "C" void kernel_launch(void* const* d_in, const int* in_sizes, int n_in,
                              void* d_out, int out_size, void* d_ws, size_t ws_size,
                              hipStream_t stream) {
    (void)in_sizes; (void)n_in; (void)out_size; (void)ws_size;
    const float* x  = (const float*)d_in[0];
    const float* Wq = (const float*)d_in[1];
    const float* bq = (const float*)d_in[2];
    const float* Wk = (const float*)d_in[3];
    const float* bk = (const float*)d_in[4];
    const float* Wv = (const float*)d_in[5];
    const float* bv = (const float*)d_in[6];

    // workspace: Q,K [b][n][c], VT [b][c][n] bf16 (4MB each) + 3 bf16 weights
    unsigned short* Q   = (unsigned short*)d_ws;
    unsigned short* K   = Q  + (size_t)2 * N_TOK * C_DIM;
    unsigned short* VT  = K  + (size_t)2 * N_TOK * C_DIM;
    unsigned short* Wqb = VT + (size_t)2 * N_TOK * C_DIM;
    unsigned short* Wkb = Wqb + (size_t)C_DIM * C_DIM;
    unsigned short* Wvb = Wkb + (size_t)C_DIM * C_DIM;

    wcvt_kernel<<<dim3(64), dim3(256), 0, stream>>>(Wq, Wqb);
    wcvt_kernel<<<dim3(64), dim3(256), 0, stream>>>(Wk, Wkb);
    wcvt_kernel<<<dim3(64), dim3(256), 0, stream>>>(Wv, Wvb);

    dim3 gProj(2 * 256), bProj(128);
    proj_kernel<<<gProj, bProj, 0, stream>>>(x, Wqb, bq, Q,  C_DIM, 1);
    proj_kernel<<<gProj, bProj, 0, stream>>>(x, Wkb, bk, K,  C_DIM, 1);
    proj_kernel<<<gProj, bProj, 0, stream>>>(x, Wvb, bv, VT, 1, N_TOK);

    const size_t lds = (size_t)16 * N_TOK * 2 + (size_t)16 * C_DIM * 2; // 136KB
    attn_kernel<<<dim3(2 * 256), dim3(256), lds, stream>>>(Q, K, VT, (float*)d_out);
}
